// VQVAE_74801150427371
// MI455X (gfx1250) — matrix operand order repaired and checked by reading, not compiled, not run on hardware
//
#include <hip/hip_runtime.h>

// ---------------------------------------------------------------------------
// VQ-VAE forward on gfx1250.
//  * all activations NHWC; K ordered (kh,kw)-major, ci-minor so every 32-wide
//    K-step is a contiguous channel run -> A fragments are 2x b128 loads
//  * conv / convT / VQ-distance as implicit GEMM on v_wmma_f32_16x16x32_f16
//  * weights & codebook pre-packed into WMMA B-fragment order
//  * per-K-step weight chunks staged to LDS with global_load_async_to_lds_b128
//    (double buffered, s_wait_asynccnt) and consumed with ds_load_b128
// ---------------------------------------------------------------------------

typedef __attribute__((ext_vector_type(16))) _Float16 v16h;
typedef __attribute__((ext_vector_type(8)))  _Float16 v8h;
typedef __attribute__((ext_vector_type(8)))  float    v8f;

#define WPB 8              // waves per block
#define TPB (32 * WPB)     // 256 threads
#define VQD_ 64
#define VQK_ 512

__device__ __forceinline__ v8f wmma_f16(v16h a, v16h b, v8f c) {
  // (neg_a, A, neg_b, B, c_mod, C, reuse_a, reuse_b)
  return __builtin_amdgcn_wmma_f32_16x16x32_f16(false, a, false, b, (short)0, c,
                                                false, false);
}

// Async copy 16B/lane from global to LDS (GVS mode: vdst=LDS addr, vaddr=off,
// saddr=base). Tracked by ASYNCcnt.
__device__ __forceinline__ void async_b128_to_lds(unsigned lds_byte,
                                                  unsigned glb_byte,
                                                  const void* base) {
  asm volatile("global_load_async_to_lds_b128 %0, %1, %2"
               :
               : "v"(lds_byte), "v"(glb_byte), "s"(base)
               : "memory");
}
__device__ __forceinline__ void wait_async0() {
  asm volatile("s_wait_asynccnt 0" ::: "memory");
}
__device__ __forceinline__ void wait_async1() {
  asm volatile("s_wait_asynccnt 1" ::: "memory");
}

// ---------------------------------------------------------------------------
// Implicit-GEMM conv / transposed-conv (NHWC).
//   M = B*Hout*Wout, N = Cout, K = KH*KW*CIN with k = (kh*KW+kw)*CIN + ci
//   Each wave: 16 x (16*NTB) output tile, A fragment reused across NTB WMMAs.
//   wtF: fragment-packed weights [(ks*NT+nt)*512 + lane*16 + j].
// ---------------------------------------------------------------------------
template <int CIN, int COUT, int KH, int KW, int STRIDE, int PAD, bool RELU, bool TRANS>
__global__ __launch_bounds__(TPB) void conv_wmma_kernel(
    const _Float16* __restrict__ in, const _Float16* __restrict__ wtF,
    const float* __restrict__ bias, _Float16* __restrict__ out16,
    float* __restrict__ out32, int NB, int Hin, int Win, int Hout, int Wout) {
  constexpr int CP = (COUT < 16) ? 16 : COUT;
  constexpr int NT = CP / 16;               // total N tiles in wtF
  constexpr int NTB = (NT < 4) ? NT : 4;    // N tiles per block
  constexpr int KTOT = CIN * KH * KW;
  constexpr int KSTEPS = (KTOT + 31) / 32;
  constexpr int CHUNK = NTB * 512;          // halves per K-step chunk

  __shared__ __align__(32) _Float16 ldsW[2][CHUNK];

  const int tid = threadIdx.x;
  const int lane = tid & 31;
  const int wave = tid >> 5;
  const int hw = Hout * Wout;
  const int Mtot = NB * hw;
  const int mBase = (blockIdx.x * WPB + wave) * 16;
  const int nt0 = blockIdx.y * NTB;

  // A-row coordinates for this lane
  const int m = mBase + (lane & 15);
  const bool mOK = (m < Mtot);
  const int mm = mOK ? m : 0;
  const int b = mm / hw;
  const int rem = mm - b * hw;
  const int oy = rem / Wout;
  const int ox = rem - oy * Wout;
  const int kbA = (lane >> 4) * 8;  // interleaved A-fragment K base

  const unsigned ldsBase = (unsigned)(uintptr_t)&ldsW[0][0];
  auto issue = [&](int ks, int buf) {
    if (tid * 8 < CHUNK)  // wave-aligned predicate (CHUNK/8 multiple of 32)
      async_b128_to_lds(ldsBase + (unsigned)(buf * (CHUNK * 2) + tid * 16),
                        (unsigned)(((ks * NT + nt0) * 512 + tid * 8) * 2), wtF);
  };

  v8f acc[NTB] = {};
  issue(0, 0);
  for (int ks = 0; ks < KSTEPS; ++ks) {
    const int cur = ks & 1;
    if (ks + 1 < KSTEPS) {  // prefetch next chunk, wait for current
      issue(ks + 1, cur ^ 1);
      wait_async1();
    } else {
      wait_async0();
    }
    __syncthreads();  // all waves' async writes visible

    // ---- A fragment ----
    v16h a;
    if constexpr ((CIN % 32) == 0) {
      // whole K-step lies in one (kh,kw) tap: vectorized NHWC channel run
      const int p = (ks * 32) / CIN;
      const int cbase = ks * 32 - p * CIN;
      const int kh = p / KW;
      const int kw = p - kh * KW;
      int iy, ix;
      bool ok;
      if constexpr (TRANS) {  // ConvTranspose2d gather: oy = iy*s + kh
        const int ty = oy - kh, tx = ox - kw;
        iy = ty / STRIDE;
        ix = tx / STRIDE;
        ok = (ty >= 0) && (tx >= 0) && (ty - iy * STRIDE == 0) &&
             (tx - ix * STRIDE == 0) && (iy < Hin) && (ix < Win);
      } else {
        iy = oy * STRIDE - PAD + kh;
        ix = ox * STRIDE - PAD + kw;
        ok = (iy >= 0) && (iy < Hin) && (ix >= 0) && (ix < Win);
      }
      ok = ok && mOK;
      const _Float16* base =
          in + (((size_t)b * Hin + iy) * Win + ix) * CIN + cbase + kbA;
      v8h lo = {}, hi = {};
      if (ok) {
        lo = *(const v8h*)base;         // K = kbA + 0..7
        hi = *(const v8h*)(base + 16);  // K = kbA + 16..23
      }
      a = __builtin_shufflevector(lo, hi, 0, 1, 2, 3, 4, 5, 6, 7, 8, 9, 10, 11,
                                  12, 13, 14, 15);
    } else {  // scalar fallback (enc1, CIN==1)
#pragma unroll
      for (int j = 0; j < 16; ++j) {
        const int ka = ks * 32 + kbA + ((j < 8) ? j : (j + 8));
        _Float16 av = (_Float16)0.f;
        if (mOK && ka < KTOT) {
          const int ci = ka % CIN;
          const int p = ka / CIN;
          const int kh = p / KW;
          const int kw = p - kh * KW;
          int iy, ix;
          bool ok;
          if constexpr (TRANS) {
            const int ty = oy - kh, tx = ox - kw;
            iy = ty / STRIDE;
            ix = tx / STRIDE;
            ok = (ty >= 0) && (tx >= 0) && (ty - iy * STRIDE == 0) &&
                 (tx - ix * STRIDE == 0) && (iy < Hin) && (ix < Win);
          } else {
            iy = oy * STRIDE - PAD + kh;
            ix = ox * STRIDE - PAD + kw;
            ok = (iy >= 0) && (iy < Hin) && (ix >= 0) && (ix < Win);
          }
          if (ok) av = in[(((size_t)b * Hin + iy) * Win + ix) * CIN + ci];
        }
        a[j] = av;
      }
    }

    // ---- NTB WMMAs reusing A, B fragments from LDS (ds_load_b128) ----
#pragma unroll
    for (int t = 0; t < NTB; ++t) {
      const v16h bf = *(const v16h*)&ldsW[cur][(t * 32 + lane) * 16];
      acc[t] = wmma_f16(a, bf, acc[t]);
    }
    __syncthreads();  // done reading before this buffer is overwritten
  }

  // ---- epilogue (NHWC stores: lanes write contiguous channels) ----
#pragma unroll
  for (int t = 0; t < NTB; ++t) {
    const int n = (nt0 + t) * 16 + (lane & 15);
    const float bv = (n < COUT) ? bias[n] : 0.f;
#pragma unroll
    for (int r = 0; r < 8; ++r) {
      const int mr = mBase + r + ((lane >> 4) << 3);
      if (mr < Mtot && n < COUT) {
        float v = acc[t][r] + bv;
        if (RELU) v = fmaxf(v, 0.f);
        const size_t o = (size_t)mr * COUT + n;
        if (out16) out16[o] = (_Float16)v;
        if (out32) out32[o] = v;
      }
    }
  }
}

// ---------------------------------------------------------------------------
// VQ nearest-code search: score(n) = ||e_n||^2 - 2 z.e_n via WMMA.
// z16 NHWC-flat [N,64] (rows contiguous); embF fragment-packed codebook.
// ---------------------------------------------------------------------------
__global__ __launch_bounds__(TPB) void vq_argmin_kernel(
    const _Float16* __restrict__ z16, const _Float16* __restrict__ embF,
    const float* __restrict__ esq, int* __restrict__ idx, int Ntot) {
  const int lane = threadIdx.x & 31;
  const int wave = threadIdx.x >> 5;
  const int mBase = (blockIdx.x * WPB + wave) * 16;
  if (mBase >= Ntot) return;  // no barriers below

  const int m = mBase + (lane & 15);
  const int mm = (m < Ntot) ? m : 0;
  const int kbA = (lane >> 4) * 8;
  const _Float16* zrow = z16 + (size_t)mm * VQD_;

  // Hoist both A fragments (K = 64 in two 32-steps); rows are contiguous.
  const v8h l0 = *(const v8h*)(zrow + kbA);
  const v8h h0 = *(const v8h*)(zrow + kbA + 16);
  const v8h l1 = *(const v8h*)(zrow + kbA + 32);
  const v8h h1 = *(const v8h*)(zrow + kbA + 48);
  const v16h a0 = __builtin_shufflevector(l0, h0, 0, 1, 2, 3, 4, 5, 6, 7, 8, 9,
                                          10, 11, 12, 13, 14, 15);
  const v16h a1 = __builtin_shufflevector(l1, h1, 0, 1, 2, 3, 4, 5, 6, 7, 8, 9,
                                          10, 11, 12, 13, 14, 15);

  float bestV[8];
  int bestI[8];
#pragma unroll
  for (int r = 0; r < 8; ++r) { bestV[r] = 3.4e38f; bestI[r] = 0; }

  for (int kt = 0; kt < VQK_ / 16; ++kt) {
    const v16h b0 = *(const v16h*)&embF[((size_t)(kt * 2 + 0) * 32 + lane) * 16];
    const v16h b1 = *(const v16h*)&embF[((size_t)(kt * 2 + 1) * 32 + lane) * 16];
    v8f acc = {};
    acc = wmma_f16(a0, b0, acc);
    acc = wmma_f16(a1, b1, acc);
    const int n = kt * 16 + (lane & 15);
    const float eq = esq[n];
#pragma unroll
    for (int r = 0; r < 8; ++r) {
      const float s = eq - 2.f * acc[r];
      if (s < bestV[r]) { bestV[r] = s; bestI[r] = n; }
    }
  }

  // 16-lane min+argmin reduction within each half-wave.
#pragma unroll
  for (int r = 0; r < 8; ++r) {
    float v = bestV[r];
    int i = bestI[r];
#pragma unroll
    for (int off = 1; off < 16; off <<= 1) {
      const float ov = __shfl_xor(v, off, 32);
      const int oi = __shfl_xor(i, off, 32);
      if (ov < v || (ov == v && oi < i)) { v = ov; i = oi; }
    }
    if ((lane & 15) == 0) {
      const int mr = mBase + r + ((lane >> 4) << 3);
      if (mr < Ntot) idx[mr] = i;
    }
  }
}

// --------------------------- small helper kernels ---------------------------
__global__ void cast_f32_f16_kernel(const float* __restrict__ in,
                                    _Float16* __restrict__ out, int n) {
  const int t = blockIdx.x * blockDim.x + threadIdx.x;
  if (t < n) out[t] = (_Float16)in[t];
}

// Pack conv/convT weights into WMMA B-fragment order with HWC K-ordering:
//   k = (kh*KW + kw)*Cin + ci
//   wtF[(ks*NT + nt)*512 + lane*16 + j] = W[k = ks*32 + 8*(lane>>4) + (j<8 ? j : j+8),
//                                          n = nt*16 + (lane&15)]
// conv  : W[k,n] = w[(n*Cin + ci)*KHW + p]   (w is OIHW [Cout,Cin,kh,kw])
// convT : W[k,n] = w[(ci*Cout + n)*KHW + p]  (w is (Cin, Cout, kh, kw))
__global__ void pack_w_frag_kernel(const float* __restrict__ w,
                                   _Float16* __restrict__ wtF, int Cout, int CP,
                                   int Cin, int KHW, int trans) {
  const int KTOT = Cin * KHW;
  const int NT = CP / 16;
  const int total = ((KTOT + 31) / 32) * NT * 512;
  const int t = blockIdx.x * blockDim.x + threadIdx.x;
  if (t >= total) return;
  const int j = t & 15;
  const int lane = (t >> 4) & 31;
  const int nt = (t >> 9) % NT;
  const int ks = (t >> 9) / NT;
  const int k = ks * 32 + ((lane >> 4) << 3) + ((j < 8) ? j : (j + 8));
  const int n = nt * 16 + (lane & 15);
  float v = 0.f;
  if (k < KTOT && n < Cout) {
    const int ci = k % Cin;
    const int p = k / Cin;
    v = trans ? w[((size_t)ci * Cout + n) * KHW + p]
              : w[((size_t)n * Cin + ci) * KHW + p];
  }
  wtF[t] = (_Float16)v;
}

__global__ void pack_emb_frag_kernel(const float* __restrict__ emb,
                                     _Float16* __restrict__ embF) {
  const int t = blockIdx.x * blockDim.x + threadIdx.x;
  if (t >= (VQK_ / 16) * 2 * 512) return;
  const int j = t & 15;
  const int lane = (t >> 4) & 31;
  const int ks = (t >> 9) & 1;
  const int kt = t >> 10;
  const int d = ks * 32 + ((lane >> 4) << 3) + ((j < 8) ? j : (j + 8));
  const int n = kt * 16 + (lane & 15);
  embF[t] = (_Float16)emb[(size_t)n * VQD_ + d];
}

__global__ void esq_kernel(const float* __restrict__ emb,
                           float* __restrict__ esq) {
  const int k = blockIdx.x * blockDim.x + threadIdx.x;
  if (k >= VQK_) return;
  float s = 0.f;
  for (int d = 0; d < VQD_; ++d) {
    const float v = emb[(size_t)k * VQD_ + d];
    s += v * v;
  }
  esq[k] = s;
}

__global__ void zero_i32_kernel(int* __restrict__ p, int n) {
  const int t = blockIdx.x * blockDim.x + threadIdx.x;
  if (t < n) p[t] = 0;
}

// z_q gather + deterministic per-block loss partials (all NHWC-flat [N,64]).
__global__ __launch_bounds__(TPB) void vq_gather_loss_kernel(
    const int* __restrict__ idx, const float* __restrict__ emb,
    const float* __restrict__ z32, _Float16* __restrict__ zq16,
    float* __restrict__ lossPart, int Ntot) {
  const int t = blockIdx.x * blockDim.x + threadIdx.x;
  float contrib = 0.f;
  if (t < Ntot * VQD_) {
    const int n = t >> 6, d = t & 63;
    const int e = idx[n];
    const float q = emb[(size_t)e * VQD_ + d];
    const float diff = q - z32[t];
    zq16[t] = (_Float16)q;  // straight-through forward value == z_q
    contrib = diff * diff;
  }
#pragma unroll
  for (int off = 16; off >= 1; off >>= 1) contrib += __shfl_down(contrib, off, 32);
  __shared__ float red[WPB];
  if ((threadIdx.x & 31) == 0) red[threadIdx.x >> 5] = contrib;
  __syncthreads();
  if (threadIdx.x == 0) {
    float s = 0.f;
    for (int w = 0; w < WPB; ++w) s += red[w];
    lossPart[blockIdx.x] = s;
  }
}

__global__ void vq_hist_kernel(const int* __restrict__ idx,
                               int* __restrict__ cnts, int Ntot) {
  const int t = blockIdx.x * blockDim.x + threadIdx.x;
  if (t < Ntot) atomicAdd(&cnts[idx[t]], 1);
}

__global__ void idx_to_float_kernel(const int* __restrict__ idx,
                                    float* __restrict__ out, int n) {
  const int t = blockIdx.x * blockDim.x + threadIdx.x;
  if (t < n) out[t] = (float)idx[t];
}

__global__ void vq_finalize_kernel(const int* __restrict__ cnts,
                                   const float* __restrict__ lossPart, int nPart,
                                   float* __restrict__ out_loss,
                                   float* __restrict__ out_perp, int Ntot) {
  __shared__ float sh[512];
  const int t = threadIdx.x;
  const float c = (float)cnts[t] / (float)Ntot;
  sh[t] = -c * logf(c + 1e-10f);
  __syncthreads();
  for (int s = 256; s > 0; s >>= 1) {
    if (t < s) sh[t] += sh[t + s];
    __syncthreads();
  }
  float perp = 0.f;
  if (t == 0) perp = expf(sh[0]);
  __syncthreads();
  float ls = 0.f;
  for (int i = t; i < nPart; i += 512) ls += lossPart[i];
  sh[t] = ls;
  __syncthreads();
  for (int s = 256; s > 0; s >>= 1) {
    if (t < s) sh[t] += sh[t + s];
    __syncthreads();
  }
  if (t == 0) {
    *out_perp = perp;
    *out_loss = sh[0] * 1.25f / (float)(Ntot * VQD_);  // (1 + BETA) * MSE
  }
}

// ---------------------------------------------------------------------------
extern "C" void kernel_launch(void* const* d_in, const int* in_sizes, int n_in,
                              void* d_out, int out_size, void* d_ws,
                              size_t ws_size, hipStream_t stream) {
  (void)in_sizes; (void)n_in; (void)out_size; (void)ws_size;
  const float* x   = (const float*)d_in[0];
  const float* ew1 = (const float*)d_in[1];  const float* eb1 = (const float*)d_in[2];
  const float* ew2 = (const float*)d_in[3];  const float* eb2 = (const float*)d_in[4];
  const float* ew3 = (const float*)d_in[5];  const float* eb3 = (const float*)d_in[6];
  const float* ew4 = (const float*)d_in[7];  const float* eb4 = (const float*)d_in[8];
  const float* emb = (const float*)d_in[9];
  const float* dw1 = (const float*)d_in[10]; const float* db1 = (const float*)d_in[11];
  const float* dw2 = (const float*)d_in[12]; const float* db2 = (const float*)d_in[13];
  const float* dw3 = (const float*)d_in[14]; const float* db3 = (const float*)d_in[15];
  const float* dw4 = (const float*)d_in[16]; const float* db4 = (const float*)d_in[17];

  float* out = (float*)d_out;
  float* out_loss = out;                       // scalar
  float* out_dec  = out + 1;                   // [64,1,103,103] (NHWC == NCHW, C=1)
  float* out_perp = out + 1 + 64 * 103 * 103;  // scalar
  float* out_idx  = out + 2 + 64 * 103 * 103;  // [64,32,32] as float

  // ---- workspace carve-up (256B aligned) ----
  size_t off = 0;
  auto A = [&](size_t bytes) -> void* {
    void* p = (char*)d_ws + off;
    off += (bytes + 255) & ~(size_t)255;
    return p;
  };
  auto fragHalves = [](int K, int CP) -> size_t {
    return (size_t)((K + 31) / 32) * (CP / 16) * 512;
  };
  _Float16* X16  = (_Float16*)A((size_t)64 * 128 * 128 * 2);  // NHWC, C=1
  _Float16* BUF0 = (_Float16*)A((size_t)64 * 128 * 103 * 103 * 2);  // A1/A3/D1/D3
  _Float16* BUF1 = (_Float16*)A((size_t)64 * 128 * 101 * 101 * 2);  // A2/D2
  _Float16* Z16  = (_Float16*)A((size_t)64 * 64 * 32 * 32 * 2);     // [N,64]
  float*    Z32  = (float*)   A((size_t)64 * 64 * 32 * 32 * 4);     // [N,64]
  _Float16* ZQ16 = (_Float16*)A((size_t)64 * 64 * 32 * 32 * 2);     // [N,64]
  _Float16* wt1  = (_Float16*)A(fragHalves(9, 128) * 2);
  _Float16* wt2  = (_Float16*)A(fragHalves(1152, 128) * 2);
  _Float16* wt3  = (_Float16*)A(fragHalves(1152, 128) * 2);
  _Float16* wt4  = (_Float16*)A(fragHalves(1152, 64) * 2);
  _Float16* wd1  = (_Float16*)A(fragHalves(2304, 128) * 2);
  _Float16* wd2  = (_Float16*)A(fragHalves(1152, 128) * 2);
  _Float16* wd3  = (_Float16*)A(fragHalves(1152, 128) * 2);
  _Float16* wd4  = (_Float16*)A(fragHalves(128, 16) * 2);
  _Float16* embF = (_Float16*)A((size_t)32 * 2 * 512 * 2);
  float*    esq  = (float*)A(VQK_ * 4);
  int*      idx  = (int*)A((size_t)65536 * 4);
  int*      cnts = (int*)A(VQK_ * 4);
  const int nLossPart = (65536 * VQD_) / TPB;  // 16384 blocks
  float* lossPart = (float*)A((size_t)nLossPart * 4);

  auto g1 = [](long n) { return dim3((unsigned)((n + 255) / 256)); };
  auto cg = [](int Mtot, int yTiles) {  // implicit-GEMM grid
    const int mtiles = (Mtot + 15) / 16;
    return dim3((unsigned)((mtiles + WPB - 1) / WPB), (unsigned)yTiles);
  };

  // ---- packing / init ----
  cast_f32_f16_kernel<<<g1(64 * 128 * 128), 256, 0, stream>>>(x, X16, 64 * 128 * 128);
  pack_w_frag_kernel<<<g1((long)fragHalves(9, 128)), 256, 0, stream>>>(ew1, wt1, 128, 128, 1, 9, 0);
  pack_w_frag_kernel<<<g1((long)fragHalves(1152, 128)), 256, 0, stream>>>(ew2, wt2, 128, 128, 128, 9, 0);
  pack_w_frag_kernel<<<g1((long)fragHalves(1152, 128)), 256, 0, stream>>>(ew3, wt3, 128, 128, 128, 9, 0);
  pack_w_frag_kernel<<<g1((long)fragHalves(1152, 64)), 256, 0, stream>>>(ew4, wt4, 64, 64, 128, 9, 0);
  pack_w_frag_kernel<<<g1((long)fragHalves(2304, 128)), 256, 0, stream>>>(dw1, wd1, 128, 128, 64, 36, 1);
  pack_w_frag_kernel<<<g1((long)fragHalves(1152, 128)), 256, 0, stream>>>(dw2, wd2, 128, 128, 128, 9, 1);
  pack_w_frag_kernel<<<g1((long)fragHalves(1152, 128)), 256, 0, stream>>>(dw3, wd3, 128, 128, 128, 9, 1);
  pack_w_frag_kernel<<<g1((long)fragHalves(128, 16)), 256, 0, stream>>>(dw4, wd4, 1, 16, 128, 1, 1);
  pack_emb_frag_kernel<<<g1(32 * 2 * 512), 256, 0, stream>>>(emb, embF);
  esq_kernel<<<g1(VQK_), 256, 0, stream>>>(emb, esq);
  zero_i32_kernel<<<g1(VQK_), 256, 0, stream>>>(cnts, VQK_);

  // ---- encoder (all NHWC) ----
  conv_wmma_kernel<1, 128, 3, 3, 2, 1, true, false><<<cg(64 * 64 * 64, 2), TPB, 0, stream>>>(
      X16, wt1, eb1, BUF0, nullptr, 64, 128, 128, 64, 64);
  conv_wmma_kernel<128, 128, 3, 3, 2, 1, true, false><<<cg(64 * 32 * 32, 2), TPB, 0, stream>>>(
      BUF0, wt2, eb2, BUF1, nullptr, 64, 64, 64, 32, 32);
  conv_wmma_kernel<128, 128, 3, 3, 1, 1, true, false><<<cg(64 * 32 * 32, 2), TPB, 0, stream>>>(
      BUF1, wt3, eb3, BUF0, nullptr, 64, 32, 32, 32, 32);
  conv_wmma_kernel<128, 64, 3, 3, 1, 1, true, false><<<cg(64 * 32 * 32, 1), TPB, 0, stream>>>(
      BUF0, wt4, eb4, Z16, Z32, 64, 32, 32, 32, 32);

  // ---- vector quantization ----
  vq_argmin_kernel<<<dim3((65536 / 16 + WPB - 1) / WPB), TPB, 0, stream>>>(
      Z16, embF, esq, idx, 65536);
  vq_gather_loss_kernel<<<g1((long)65536 * VQD_), TPB, 0, stream>>>(
      idx, emb, Z32, ZQ16, lossPart, 65536);
  vq_hist_kernel<<<g1(65536), 256, 0, stream>>>(idx, cnts, 65536);
  idx_to_float_kernel<<<g1(65536), 256, 0, stream>>>(idx, out_idx, 65536);
  vq_finalize_kernel<<<1, 512, 0, stream>>>(cnts, lossPart, nLossPart, out_loss,
                                            out_perp, 65536);

  // ---- decoder (all NHWC; dec4 output C=1 -> NCHW-identical) ----
  conv_wmma_kernel<64, 128, 6, 6, 3, 0, true, true><<<cg(64 * 99 * 99, 2), TPB, 0, stream>>>(
      ZQ16, wd1, db1, BUF0, nullptr, 64, 32, 32, 99, 99);
  conv_wmma_kernel<128, 128, 3, 3, 1, 0, true, true><<<cg(64 * 101 * 101, 2), TPB, 0, stream>>>(
      BUF0, wd2, db2, BUF1, nullptr, 64, 99, 99, 101, 101);
  conv_wmma_kernel<128, 128, 3, 3, 1, 0, true, true><<<cg(64 * 103 * 103, 2), TPB, 0, stream>>>(
      BUF1, wd3, db3, BUF0, nullptr, 64, 101, 101, 103, 103);
  conv_wmma_kernel<128, 1, 1, 1, 1, 0, false, true><<<cg(64 * 103 * 103, 1), TPB, 0, stream>>>(
      BUF0, wd4, db4, nullptr, out_dec, 64, 103, 103, 103, 103);
}
